// GatedTorsion_36867999269029
// MI455X (gfx1250) — matrix operand order repaired and checked by reading, not compiled, not run on hardware
//
#include <hip/hip_runtime.h>
#include <hip/hip_bf16.h>
#include <math.h>

// ---------------- problem constants ----------------
#define REP   128
#define XDIM  512      // 4*REP
#define BT    256      // between_feats
#define NPER  6
#define NT    500000
#define NN    200000

#define WAVES_PER_WG 4          // 128 threads; 96KB LDS/WG -> 3 WGs per 320KB WGP
#define LDS_HALFS_PER_WAVE 12288

typedef __attribute__((ext_vector_type(16))) _Float16 v16h;
typedef __attribute__((ext_vector_type(8)))  _Float16 v8h;
typedef __attribute__((ext_vector_type(8)))  float    v8f;

union V16U { v16h v; v8h h[2]; };

// f16 weight offsets inside d_ws (units: halfs)
#define OFF_SW   0         // 256 x 512
#define OFF_TRW  131072    // 256 x 256
#define OFF_TW1  196608    // 256 x 256
#define OFF_TW2  262144    // 16  x 256 (rows 6..15 zero)
#define OFF_CRW  266240    // 256 x 256
#define OFF_CW1  331776    // 256 x 256
#define OFF_CW2  397312    // 16  x 256 (rows 6..15 zero)

__device__ __forceinline__ float eluf(float v) {
  return v > 0.f ? v : (__expf(v) - 1.f);
}

// D = bias + Xtile(16xK) * W16rows(Kx16) via WMMA f16->f32.
// Two interleaved accumulators: dependent v_wmma are 2 apart, not back-to-back
// (f16 WMMA->WMMA RAW hazard is 5 slots per ISA 7.12.1).
// A fragment: lane (grp,m): halfs K = c*32 + grp*8 .. +7 and +16..+23 (ISA A layout)
// B fragment: lane (grp,n): halfs K = c*32 + grp*8 .. +7 and +16..+23 of weight row n (same K order as A)
template<int KCH>
__device__ __forceinline__ v8f gemm_acc(const _Float16* __restrict__ xbase, int xstride,
                                        const _Float16* __restrict__ wrow,
                                        v8f acc, int m, int grp) {
  v8f a0 = acc;
  v8f a1;
#pragma unroll
  for (int r = 0; r < 8; ++r) a1[r] = 0.f;
#pragma unroll 4
  for (int c = 0; c < KCH; ++c) {
    const _Float16* pa = xbase + m * xstride + c * 32 + grp * 8;
    V16U ua;
    ua.h[0] = *(const v8h*)(pa);
    ua.h[1] = *(const v8h*)(pa + 16);
    const _Float16* pb = wrow + c * 32 + grp * 8;
    V16U ub;
    ub.h[0] = *(const v8h*)(pb);
    ub.h[1] = *(const v8h*)(pb + 16);
    if (c & 1)
      a1 = __builtin_amdgcn_wmma_f32_16x16x32_f16(false, ua.v, false, ub.v,
                                                  (short)0, a1, false, false);
    else
      a0 = __builtin_amdgcn_wmma_f32_16x16x32_f16(false, ua.v, false, ub.v,
                                                  (short)0, a0, false, false);
  }
  return a0 + a1;
}

// One head: y = x + elu(x@rw.T+rb); z = elu(y@w1.T+b1); out = z@w2.T+b2 (NPER cols)
__device__ __forceinline__ v8f head_eval(const _Float16* __restrict__ X1,
                                         _Float16* __restrict__ Y,
                                         _Float16* __restrict__ Z,
                                         const _Float16* __restrict__ rw, const float* __restrict__ rb,
                                         const _Float16* __restrict__ w1, const float* __restrict__ b1,
                                         const _Float16* __restrict__ w2, const float* __restrict__ b2,
                                         int m, int grp, int lane) {
  const int col = lane & 15;

  // residual layer
  for (int nt = 0; nt < 16; ++nt) {
    const int n = nt * 16 + col;
    const float bb = rb[n];
    v8f acc;
#pragma unroll
    for (int r = 0; r < 8; ++r) acc[r] = bb;
    acc = gemm_acc<8>(X1, BT, rw + n * BT, acc, m, grp);
#pragma unroll
    for (int r = 0; r < 8; ++r) {
      const int row = grp * 8 + r;
      const float xv = (float)X1[row * BT + n];
      Y[row * BT + n] = (_Float16)(xv + eluf(acc[r]));
    }
  }
  __syncthreads();

  // layer 1
  for (int nt = 0; nt < 16; ++nt) {
    const int n = nt * 16 + col;
    const float bb = b1[n];
    v8f acc;
#pragma unroll
    for (int r = 0; r < 8; ++r) acc[r] = bb;
    acc = gemm_acc<8>(Y, BT, w1 + n * BT, acc, m, grp);
#pragma unroll
    for (int r = 0; r < 8; ++r) {
      const int row = grp * 8 + r;
      Z[row * BT + n] = (_Float16)eluf(acc[r]);
    }
  }
  __syncthreads();

  // layer 2: single 16-wide N tile (rows >= NPER of w2 are zero-padded)
  const float bb = (col < NPER) ? b2[col] : 0.f;
  v8f acc;
#pragma unroll
  for (int r = 0; r < 8; ++r) acc[r] = bb;
  acc = gemm_acc<8>(Z, BT, w2 + col * BT, acc, m, grp);
  return acc;
}

__global__ void __launch_bounds__(WAVES_PER_WG * 32) gated_torsion_main(
    const float* __restrict__ h, const int* __restrict__ idxs,
    const _Float16* __restrict__ wsh,
    const float* __restrict__ sb,
    const float* __restrict__ t_rb, const float* __restrict__ t_b1, const float* __restrict__ t_b2,
    const float* __restrict__ c_rb, const float* __restrict__ c_b1, const float* __restrict__ c_b2,
    float* __restrict__ out_score, float* __restrict__ out_k) {
  extern __shared__ _Float16 smem[];
  const int lane  = threadIdx.x & 31;
  const int wave  = threadIdx.x >> 5;
  const int m     = lane & 15;
  const int grp   = lane >> 4;
  const int tile  = blockIdx.x * WAVES_PER_WG + wave;   // 16-torsion tile per wave
  const int tbase = tile * 16;

  _Float16* Xg = smem + wave * LDS_HALFS_PER_WAVE;  // 16x512 gather; later reused as Y/Z
  _Float16* X1 = Xg + 8192;                         // 16x256 post-shared activations

  // ---- gather: 16 torsions x 4 nodes x 128 feats, f32 -> f16 into LDS ----
  for (int i = lane; i < 2048; i += 32) {          // 2048 float4 per tile
    const int row = i >> 7;                        // 128 float4 per torsion row
    const int seg = i & 127;
    int t = tbase + row; if (t >= NT) t = NT - 1;
    const int node = idxs[t * 4 + (seg >> 5)];     // 32 float4 per node
    const float4 f = *(const float4*)(h + (size_t)node * REP + (size_t)(seg & 31) * 4);
    _Float16* dst = Xg + row * XDIM + seg * 4;
    dst[0] = (_Float16)f.x; dst[1] = (_Float16)f.y;
    dst[2] = (_Float16)f.z; dst[3] = (_Float16)f.w;
  }
  __syncthreads();

  // ---- shared layer: X1 = elu(X @ sw.T + sb), K = 512 ----
  const _Float16* swh = wsh + OFF_SW;
  for (int nt = 0; nt < 16; ++nt) {
    const int n = nt * 16 + m;
    const float bb = sb[n];
    v8f acc;
#pragma unroll
    for (int r = 0; r < 8; ++r) acc[r] = bb;
    acc = gemm_acc<16>(Xg, XDIM, swh + n * XDIM, acc, m, grp);
#pragma unroll
    for (int r = 0; r < 8; ++r) {
      const int row = grp * 8 + r;
      X1[row * BT + n] = (_Float16)eluf(acc[r]);
    }
  }
  __syncthreads();

  _Float16* Y = Xg;          // recycle gather region
  _Float16* Z = Xg + 4096;

  const v8f score = head_eval(X1, Y, Z, wsh + OFF_CRW, c_rb, wsh + OFF_CW1, c_b1,
                              wsh + OFF_CW2, c_b2, m, grp, lane);
  __syncthreads();
  const v8f coef  = head_eval(X1, Y, Z, wsh + OFF_TRW, t_rb, wsh + OFF_TW1, t_b1,
                              wsh + OFF_TW2, t_b2, m, grp, lane);

  // ---- epilogue: score; k = coeffs * 1e-3 * sigmoid(score) ----
  const int col = lane & 15;
  if (col < NPER) {
#pragma unroll
    for (int r = 0; r < 8; ++r) {
      const int row = grp * 8 + r;
      const int t = tbase + row;
      if (t < NT) {
        const float s = score[r];
        out_score[(size_t)t * NPER + col] = s;
        const float g = 1.f / (1.f + __expf(-s));
        out_k[(size_t)t * NPER + col] = coef[r] * 1e-3f * g;
      }
    }
  }
}

// ---------------- weight prep: f32 -> f16 into workspace ----------------
__global__ void cvt_f32_f16(const float* __restrict__ s, _Float16* __restrict__ d, int n) {
  const int i = blockIdx.x * 256 + threadIdx.x;
  if (i < n) d[i] = (_Float16)s[i];
}

__global__ void cvt_pad_w2(const float* __restrict__ s, _Float16* __restrict__ d) {
  const int i = blockIdx.x * 256 + threadIdx.x;   // 16*256 elements
  if (i < 16 * BT) {
    const int r = i >> 8, c = i & 255;
    d[i] = (r < NPER) ? (_Float16)s[r * BT + c] : (_Float16)0.f;
  }
}

extern "C" void kernel_launch(void* const* d_in, const int* in_sizes, int n_in,
                              void* d_out, int out_size, void* d_ws, size_t ws_size,
                              hipStream_t stream) {
  (void)in_sizes; (void)n_in; (void)out_size; (void)ws_size;

  const float* h    = (const float*)d_in[0];
  const int*   idxs = (const int*)d_in[1];
  const float* sw   = (const float*)d_in[2];
  const float* sb   = (const float*)d_in[3];
  const float* t_rw = (const float*)d_in[4];
  const float* t_rb = (const float*)d_in[5];
  const float* t_w1 = (const float*)d_in[6];
  const float* t_b1 = (const float*)d_in[7];
  const float* t_w2 = (const float*)d_in[8];
  const float* t_b2 = (const float*)d_in[9];
  const float* c_rw = (const float*)d_in[10];
  const float* c_rb = (const float*)d_in[11];
  const float* c_w1 = (const float*)d_in[12];
  const float* c_b1 = (const float*)d_in[13];
  const float* c_w2 = (const float*)d_in[14];
  const float* c_b2 = (const float*)d_in[15];

  _Float16* wsh = (_Float16*)d_ws;

  cvt_f32_f16<<<(131072 + 255) / 256, 256, 0, stream>>>(sw,   wsh + OFF_SW,  131072);
  cvt_f32_f16<<<(65536  + 255) / 256, 256, 0, stream>>>(t_rw, wsh + OFF_TRW, 65536);
  cvt_f32_f16<<<(65536  + 255) / 256, 256, 0, stream>>>(t_w1, wsh + OFF_TW1, 65536);
  cvt_pad_w2 <<<16, 256, 0, stream>>>(t_w2, wsh + OFF_TW2);
  cvt_f32_f16<<<(65536  + 255) / 256, 256, 0, stream>>>(c_rw, wsh + OFF_CRW, 65536);
  cvt_f32_f16<<<(65536  + 255) / 256, 256, 0, stream>>>(c_w1, wsh + OFF_CW1, 65536);
  cvt_pad_w2 <<<16, 256, 0, stream>>>(c_w2, wsh + OFF_CW2);

  float* out_score = (float*)d_out;
  float* out_k     = out_score + (size_t)NT * NPER;

  const int nTiles = (NT + 15) / 16;                       // 31250 wave-tiles
  const int nWG    = (nTiles + WAVES_PER_WG - 1) / WAVES_PER_WG;
  const size_t shmem = (size_t)WAVES_PER_WG * LDS_HALFS_PER_WAVE * sizeof(_Float16); // 96 KB

  (void)hipFuncSetAttribute((const void*)gated_torsion_main,
                            hipFuncAttributeMaxDynamicSharedMemorySize, (int)shmem);
  gated_torsion_main<<<nWG, WAVES_PER_WG * 32, shmem, stream>>>(h, idxs, wsh, sb,
                                                                t_rb, t_b1, t_b2,
                                                                c_rb, c_b1, c_b2,
                                                                out_score, out_k);
}
